// QGM_Transformer_Decoder_13245679141125
// MI455X (gfx1250) — compile-verified, matrix operand order checked by reading, not run on hardware
//
#include <hip/hip_runtime.h>
#include <math.h>

#define B_  64
#define S_  256
#define D_  1024
#define T_  24
#define A_  128
#define C_  64
#define TB_ 16
#define NEGBIG (-1e9f)

typedef __bf16 bf16_t;
typedef __attribute__((ext_vector_type(16))) __bf16 bf16x16;
typedef __attribute__((ext_vector_type(8)))  __bf16 bf16x8;
typedef __attribute__((ext_vector_type(8)))  float  floatx8;

// ---------------------------------------------------------------------------
// Conversion / setup kernels
// ---------------------------------------------------------------------------
__global__ void k_cvt_bf16(const float* __restrict__ src, bf16_t* __restrict__ dst, int n) {
  int i = blockIdx.x * blockDim.x + threadIdx.x;
  int stride = gridDim.x * blockDim.x;
  for (; i < n; i += stride) dst[i] = (bf16_t)src[i];
}

// Wt[n*K + k] = W[k*N + n]  (store B^T so WMMA B-fragments are K-contiguous)
__global__ void k_transpose_cvt(const float* __restrict__ W, bf16_t* __restrict__ Wt,
                                int K, int N) {
  long long i = blockIdx.x * (long long)blockDim.x + threadIdx.x;
  long long total = (long long)K * N;
  long long stride = gridDim.x * (long long)blockDim.x;
  for (; i < total; i += stride) {
    int n = (int)(i / K); int k = (int)(i % K);
    Wt[i] = (bf16_t)W[(long long)k * N + n];
  }
}

// Wcat^T[n][0:3072] = W_ih[n][:], Wcat^T[n][3072:4096] = W_hh[n][:]
__global__ void k_build_wcat(const float* __restrict__ Wih, const float* __restrict__ Whh,
                             bf16_t* __restrict__ Wcat) {
  long long i = blockIdx.x * (long long)blockDim.x + threadIdx.x;
  long long total = 4096LL * 4096LL;
  long long stride = gridDim.x * (long long)blockDim.x;
  for (; i < total; i += stride) {
    int n = (int)(i >> 12); int k = (int)(i & 4095);
    float v = (k < 3072) ? Wih[(long long)n * 3072 + k]
                         : Whh[(long long)n * 1024 + (k - 3072)];
    Wcat[i] = (bf16_t)v;
  }
}

__global__ void k_setup(const float* __restrict__ init_h, const float* __restrict__ init_c,
                        const float* __restrict__ b_ih, const float* __restrict__ b_hh,
                        float* __restrict__ h, float* __restrict__ c,
                        bf16_t* __restrict__ h_bf, float* __restrict__ bsum,
                        float* __restrict__ loss) {
  int i = blockIdx.x * blockDim.x + threadIdx.x;
  if (i < B_ * D_) { float hv = init_h[i]; h[i] = hv; c[i] = init_c[i]; h_bf[i] = (bf16_t)hv; }
  if (i < 4 * D_) bsum[i] = b_ih[i] + b_hh[i];
  if (i == 0) loss[0] = 0.f;
}

// ---------------------------------------------------------------------------
// Async global -> LDS copy (CDNA5 GLOBAL_LOAD_ASYNC_TO_LDS_B128, ASYNCcnt).
// VDST = per-lane LDS byte address, VADDR = per-lane 64-bit global address.
// ---------------------------------------------------------------------------
__device__ __forceinline__ void async_copy_b128(void* lds_ptr, const void* gptr) {
  uint32_t lds_addr = (uint32_t)(uintptr_t)lds_ptr;
  asm volatile("global_load_async_to_lds_b128 %0, %1, off"
               :: "v"(lds_addr), "v"(gptr) : "memory");
}

// ---------------------------------------------------------------------------
// WMMA bf16 GEMM: C[M,N] = A[M,K](bf16 row-major) * Bt[N,K](bf16, = B^T) + bias[N]
// Requires M % 64 == 0, N % 128 == 0, K % 64 == 0 (all call sites satisfy this).
// Block: 256 threads = 8 wave32 waves, tile 64(M) x 128(N), K staged in 64-chunks,
// double-buffered LDS with async-to-LDS loads overlapped against WMMA compute.
// ---------------------------------------------------------------------------
__global__ __launch_bounds__(256) void k_gemm_bf16(
    const bf16_t* __restrict__ A, const bf16_t* __restrict__ Bt,
    const float* __restrict__ bias, float* __restrict__ Cout,
    int M, int N, int K) {
  __shared__ __align__(32) bf16_t ldsA[2][64 * 64];
  __shared__ __align__(32) bf16_t ldsB[2][128 * 64];
  const int tid  = threadIdx.x;
  const int lane = tid & 31;
  const int wave = tid >> 5;
  const int wm = wave >> 1;     // 0..3 : which 16-row strip
  const int wn = wave & 1;      // 0..1 : which 64-col half
  const int bm = blockIdx.y * 64;
  const int bn = blockIdx.x * 128;

  // Each thread stages 2 x 16B of A and 4 x 16B of B per K-chunk (6 async ops/wave).
  auto issue_chunk = [&](int kt, int buf) {
    const int k0 = kt << 6;
#pragma unroll
    for (int i = 0; i < 2; ++i) {
      int idx = tid + i * 256;                 // 512 vec8 of A tile
      int r = idx >> 3, cs = (idx & 7) << 3;
      async_copy_b128(&ldsA[buf][r * 64 + cs],
                      A + (long long)(bm + r) * K + k0 + cs);
    }
#pragma unroll
    for (int i = 0; i < 4; ++i) {
      int idx = tid + i * 256;                 // 1024 vec8 of B tile
      int r = idx >> 3, cs = (idx & 7) << 3;
      async_copy_b128(&ldsB[buf][r * 64 + cs],
                      Bt + (long long)(bn + r) * K + k0 + cs);
    }
  };

  const int nK = K >> 6;
  floatx8 acc[4] = {};
  issue_chunk(0, 0);
  for (int kt = 0; kt < nK; ++kt) {
    const int cur = kt & 1;
    if (kt + 1 < nK) {
      issue_chunk(kt + 1, cur ^ 1);
      // 6 newest async ops are the next chunk; wait for everything older.
      asm volatile("s_wait_asynccnt 0x6" ::: "memory");
    } else {
      asm volatile("s_wait_asynccnt 0x0" ::: "memory");
    }
    __syncthreads();
    const bf16_t* la = &ldsA[cur][0];
    const bf16_t* lb = &ldsB[cur][0];
#pragma unroll
    for (int kk = 0; kk < 64; kk += 32) {
      // A fragment: lanes 0-15 hold K=kk..kk+15, lanes 16-31 hold K=kk+16..kk+31
      const int lks = kk + ((lane >> 4) << 4);
      bf16x16 afrag = *(const bf16x16*)(la + (wm * 16 + (lane & 15)) * 64 + lks);
      bf16x16 bfr[4];
#pragma unroll
      for (int j = 0; j < 4; ++j)
        bfr[j] = *(const bf16x16*)(lb + (wn * 64 + j * 16 + (lane & 15)) * 64 + lks);
#pragma unroll
      for (int j = 0; j < 4; ++j)
        acc[j] = __builtin_amdgcn_wmma_f32_16x16x32_bf16(
            false, afrag, false, bfr[j], (short)0, acc[j], false, false);
    }
    __syncthreads();
  }
  // Epilogue per ISA C/D layout: VGPR v -> M = v (lanes 0-15) or v+8 (lanes 16-31)
#pragma unroll
  for (int j = 0; j < 4; ++j) {
    int gn = bn + wn * 64 + j * 16 + (lane & 15);
    float bv = bias ? bias[gn] : 0.f;
#pragma unroll
    for (int v = 0; v < 8; ++v) {
      int gm = bm + wm * 16 + v + ((lane >> 4) << 3);
      Cout[(long long)gm * N + gn] = acc[j][v] + bv;
    }
  }
}

// ---------------------------------------------------------------------------
// Attention + pack: one block per batch row (256 threads, S_ == 256)
// scores -> softmax -> ctx, then write xcat[b] = bf16([prev | node | ctx | h])
// ---------------------------------------------------------------------------
__device__ __forceinline__ float block_reduce_max(float v, float* red) {
  int tid = threadIdx.x;
  red[tid] = v; __syncthreads();
  for (int off = 128; off > 0; off >>= 1) {
    if (tid < off) red[tid] = fmaxf(red[tid], red[tid + off]);
    __syncthreads();
  }
  float r = red[0]; __syncthreads(); return r;
}
__device__ __forceinline__ float block_reduce_sum(float v, float* red) {
  int tid = threadIdx.x;
  red[tid] = v; __syncthreads();
  for (int off = 128; off > 0; off >>= 1) {
    if (tid < off) red[tid] += red[tid + off];
    __syncthreads();
  }
  float r = red[0]; __syncthreads(); return r;
}

__global__ __launch_bounds__(256) void k_attn_pack(
    const float* __restrict__ memory, const float* __restrict__ q,
    const int* __restrict__ mmask, const float* __restrict__ h,
    const float* __restrict__ prev_t, const float* __restrict__ node_t,
    bf16_t* __restrict__ xcat) {
  __shared__ float sw[S_];
  __shared__ float red[256];
  const int b = blockIdx.x, tid = threadIdx.x;
  const float4* m4 = (const float4*)(memory + ((long long)b * S_ + tid) * D_);
  const float4* q4 = (const float4*)(q + (long long)b * D_);
  float s = 0.f;
  for (int i = 0; i < D_ / 4; ++i) {
    float4 a = m4[i], w = q4[i];
    s += a.x * w.x + a.y * w.y + a.z * w.z + a.w * w.w;
  }
  if (mmask[b * S_ + tid] == 1) s += NEGBIG;
  float mx = block_reduce_max(s, red);
  float e = expf(s - mx);
  float sum = block_reduce_sum(e, red);
  sw[tid] = e / sum;
  __syncthreads();
  // ctx dims tid*4 .. tid*4+3
  float c0 = 0, c1 = 0, c2 = 0, c3 = 0;
  for (int si = 0; si < S_; ++si) {
    float w = sw[si];
    float4 m = *(const float4*)(memory + ((long long)b * S_ + si) * D_ + tid * 4);
    c0 += w * m.x; c1 += w * m.y; c2 += w * m.z; c3 += w * m.w;
  }
  bf16_t* xr = xcat + (long long)b * 4096;
  for (int i = tid; i < D_; i += 256) {
    xr[i]          = (bf16_t)prev_t[(long long)b * D_ + i];
    xr[D_ + i]     = (bf16_t)node_t[(long long)b * D_ + i];
    xr[3 * D_ + i] = (bf16_t)h[(long long)b * D_ + i];
  }
  xr[2 * D_ + tid * 4 + 0] = (bf16_t)c0;
  xr[2 * D_ + tid * 4 + 1] = (bf16_t)c1;
  xr[2 * D_ + tid * 4 + 2] = (bf16_t)c2;
  xr[2 * D_ + tid * 4 + 3] = (bf16_t)c3;
}

// ---------------------------------------------------------------------------
// LSTM pointwise (gates already include b_ih + b_hh from the GEMM epilogue)
// ---------------------------------------------------------------------------
__global__ __launch_bounds__(256) void k_lstm(const float* __restrict__ gates,
                                              float* __restrict__ h, float* __restrict__ c,
                                              bf16_t* __restrict__ h_bf) {
  int b = blockIdx.x, tid = threadIdx.x;
  const float* g = gates + (long long)b * 4 * D_;
  for (int d = tid; d < D_; d += 256) {
    float ig = g[d], fg = g[D_ + d], gg = g[2 * D_ + d], og = g[3 * D_ + d];
    float si = 1.f / (1.f + expf(-ig));
    float sf = 1.f / (1.f + expf(-fg));
    float so = 1.f / (1.f + expf(-og));
    long long idx = (long long)b * D_ + d;
    float cn = sf * c[idx] + si * tanhf(gg);
    float hn = so * tanhf(cn);
    c[idx] = cn; h[idx] = hn; h_bf[idx] = (bf16_t)hn;
  }
}

// ---------------------------------------------------------------------------
// Scoring heads + log-softmax + gold gather -> accumulate -ll into loss
// ---------------------------------------------------------------------------
__global__ __launch_bounds__(256) void k_score(
    const float* __restrict__ act_src, const float* __restrict__ col_src,
    const float* __restrict__ tab_src, const float* __restrict__ outv,
    const int* __restrict__ cmask, const int* __restrict__ tmask,
    const int* __restrict__ ga, const int* __restrict__ gc, const int* __restrict__ gt,
    float* __restrict__ loss) {
  __shared__ float sc[256];
  int b = blockIdx.x, tid = threadIdx.x;
  const float* ob = outv + (long long)b * D_;
  float v = 0.f;
  const float* src = nullptr;
  float biasadd = 0.f;
  if (tid < A_) {
    src = act_src + (long long)tid * D_;
  } else if (tid < A_ + C_) {
    int ci = tid - A_;
    src = col_src + ((long long)b * C_ + ci) * D_;
    if (cmask[b * C_ + ci] == 1) biasadd = NEGBIG;
  } else if (tid < A_ + C_ + TB_) {
    int ti = tid - A_ - C_;
    src = tab_src + ((long long)b * TB_ + ti) * D_;
    if (tmask[b * TB_ + ti] == 1) biasadd = NEGBIG;
  }
  if (src) {
    const float4* s4 = (const float4*)src;
    const float4* o4 = (const float4*)ob;
    for (int i = 0; i < D_ / 4; ++i) {
      float4 a = s4[i], w = o4[i];
      v += a.x * w.x + a.y * w.y + a.z * w.z + a.w * w.w;
    }
    v += biasadd;
  }
  sc[tid] = v;
  __syncthreads();
  if (tid == 0) {
    float ll = 0.f;
    int offs[3]  = {0, A_, A_ + C_};
    int lens[3]  = {A_, C_, TB_};
    int golds[3] = {ga[b], gc[b], gt[b]};
    for (int hI = 0; hI < 3; ++hI) {
      float m = -3.4e38f;
      for (int i = 0; i < lens[hI]; ++i) m = fmaxf(m, sc[offs[hI] + i]);
      float s = 0.f;
      for (int i = 0; i < lens[hI]; ++i) s += expf(sc[offs[hI] + i] - m);
      ll += sc[offs[hI] + golds[hI]] - m - logf(s);
    }
    atomicAdd(loss, -ll);
  }
}

// ---------------------------------------------------------------------------
// Host launch
// ---------------------------------------------------------------------------
extern "C" void kernel_launch(void* const* d_in, const int* in_sizes, int n_in,
                              void* d_out, int out_size, void* d_ws, size_t ws_size,
                              hipStream_t stream) {
  (void)in_sizes; (void)n_in; (void)out_size; (void)ws_size;
  const float* memory   = (const float*)d_in[0];
  const float* enc_col  = (const float*)d_in[1];
  const float* enc_tab  = (const float*)d_in[2];
  const float* init_h   = (const float*)d_in[3];
  const float* init_c   = (const float*)d_in[4];
  const float* prev_emb = (const float*)d_in[5];
  const float* node_emb = (const float*)d_in[6];
  const float* W_att    = (const float*)d_in[7];
  const float* b_att    = (const float*)d_in[8];
  const float* W_tgt    = (const float*)d_in[9];
  const float* b_tgt    = (const float*)d_in[10];
  const float* W_out    = (const float*)d_in[11];
  const float* b_out    = (const float*)d_in[12];
  const float* W_ih     = (const float*)d_in[13];
  const float* W_hh     = (const float*)d_in[14];
  const float* b_ih     = (const float*)d_in[15];
  const float* b_hh     = (const float*)d_in[16];
  const float* act_emb  = (const float*)d_in[17];
  const int*   mem_mask = (const int*)d_in[18];
  const int*   col_mask = (const int*)d_in[19];
  const int*   tab_mask = (const int*)d_in[20];
  const int*   gold_a   = (const int*)d_in[21];
  const int*   gold_c   = (const int*)d_in[22];
  const int*   gold_t   = (const int*)d_in[23];
  float* loss = (float*)d_out;

  char* w = (char*)d_ws;
  auto alloc = [&](size_t bytes) -> void* {
    void* p = (void*)w;
    w += (bytes + 255) & ~(size_t)255;
    return p;
  };
  bf16_t* Wcat   = (bf16_t*)alloc(4096ULL * 4096 * 2);  // [N=4096][K=4096] (== B^T)
  bf16_t* WtgtT  = (bf16_t*)alloc(1024ULL * 1024 * 2);
  bf16_t* WoutT  = (bf16_t*)alloc(1024ULL * 1024 * 2);
  bf16_t* WattT  = (bf16_t*)alloc(1024ULL * 1024 * 2);
  bf16_t* act_bf = (bf16_t*)alloc(128ULL  * 1024 * 2);
  bf16_t* col_bf = (bf16_t*)alloc(4096ULL * 1024 * 2);
  bf16_t* tab_bf = (bf16_t*)alloc(1024ULL * 1024 * 2);
  float*  act_src = (float*)alloc(128ULL  * 1024 * 4);
  float*  col_src = (float*)alloc(4096ULL * 1024 * 4);
  float*  tab_src = (float*)alloc(1024ULL * 1024 * 4);
  float*  h     = (float*)alloc((size_t)B_ * D_ * 4);
  float*  c     = (float*)alloc((size_t)B_ * D_ * 4);
  bf16_t* h_bf  = (bf16_t*)alloc((size_t)B_ * D_ * 2);
  float*  q     = (float*)alloc((size_t)B_ * D_ * 4);
  float*  gates = (float*)alloc((size_t)B_ * 4 * D_ * 4);
  float*  outv  = (float*)alloc((size_t)B_ * D_ * 4);
  bf16_t* xcat  = (bf16_t*)alloc((size_t)B_ * 4096 * 2);
  float*  bsum  = (float*)alloc((size_t)4 * D_ * 4);

  // One-time setup + weight conversion
  k_setup<<<(B_ * D_ + 255) / 256, 256, 0, stream>>>(init_h, init_c, b_ih, b_hh,
                                                     h, c, h_bf, bsum, loss);
  k_build_wcat<<<2048, 256, 0, stream>>>(W_ih, W_hh, Wcat);
  k_transpose_cvt<<<1024, 256, 0, stream>>>(W_tgt, WtgtT, 1024, 1024);
  k_transpose_cvt<<<1024, 256, 0, stream>>>(W_out, WoutT, 1024, 1024);
  k_transpose_cvt<<<1024, 256, 0, stream>>>(W_att, WattT, 1024, 1024);
  k_cvt_bf16<<<512,  256, 0, stream>>>(act_emb, act_bf, 128 * 1024);
  k_cvt_bf16<<<2048, 256, 0, stream>>>(enc_col, col_bf, 4096 * 1024);
  k_cvt_bf16<<<1024, 256, 0, stream>>>(enc_tab, tab_bf, 1024 * 1024);

  // Hoisted att_affine heads (WMMA GEMMs)
  k_gemm_bf16<<<dim3(8, 2),  256, 0, stream>>>(act_bf, WattT, b_att, act_src, 128,  1024, 1024);
  k_gemm_bf16<<<dim3(8, 64), 256, 0, stream>>>(col_bf, WattT, b_att, col_src, 4096, 1024, 1024);
  k_gemm_bf16<<<dim3(8, 16), 256, 0, stream>>>(tab_bf, WattT, b_att, tab_src, 1024, 1024, 1024);

  // Sequential decode
  for (int t = 0; t < T_; ++t) {
    const float* prev_t = prev_emb + (long long)t * B_ * D_;
    const float* node_t = node_emb + (long long)t * B_ * D_;
    k_gemm_bf16<<<dim3(8, 1), 256, 0, stream>>>(h_bf, WtgtT, b_tgt, q, 64, 1024, 1024);
    k_attn_pack<<<B_, 256, 0, stream>>>(memory, q, mem_mask, h, prev_t, node_t, xcat);
    k_gemm_bf16<<<dim3(32, 1), 256, 0, stream>>>(xcat, Wcat, bsum, gates, 64, 4096, 4096);
    k_lstm<<<B_, 256, 0, stream>>>(gates, h, c, h_bf);
    k_gemm_bf16<<<dim3(8, 1), 256, 0, stream>>>(h_bf, WoutT, b_out, outv, 64, 1024, 1024);
    k_score<<<B_, 256, 0, stream>>>(act_src, col_src, tab_src, outv, col_mask, tab_mask,
                                    gold_a + t * B_, gold_c + t * B_, gold_t + t * B_, loss);
  }
}
